// SelfAttention_25383256720199
// MI455X (gfx1250) — matrix operand order repaired and checked by reading, not compiled, not run on hardware
//
#include <hip/hip_runtime.h>
#include <hip/hip_bf16.h>

typedef __attribute__((ext_vector_type(16))) __bf16 v16bf;
typedef __attribute__((ext_vector_type(8)))  __bf16 v8bf;
typedef __attribute__((ext_vector_type(8)))  float  v8f;
typedef __attribute__((ext_vector_type(4)))  float  v4f;

#define WMMA_BF16(A, B, C) \
  __builtin_amdgcn_wmma_f32_16x16x32_bf16(false, (A), false, (B), (short)0, (C), false, false)

// ---------------------------------------------------------------------------
// Fragment loaders (CDNA5 WMMA bf16 layouts, CDNA5 ISA §7.12.2)
// A (16x32, row-major source, ld in elements):
//   lane<16 : elems 0..7 = K0..7   , elems 8..15 = K16..23   (row = lane)
//   lane>=16: elems 0..7 = K8..15  , elems 8..15 = K24..31   (row = lane-16)
// B (32x16; source stored K-contiguous per output column, i.e. transposed):
//   lane<16 : elems 0..7 = K0..7   , elems 8..15 = K16..23   (col = lane)
//   lane>=16: elems 0..7 = K8..15  , elems 8..15 = K24..31   (col = lane-16)
// ---------------------------------------------------------------------------
static __device__ __forceinline__ v16bf load_frag_a(const __bf16* __restrict__ base, int ld) {
  const int lane = threadIdx.x & 31;
  const int row  = lane & 15;
  const int half = lane >> 4;
  const __bf16* p = base + (size_t)row * ld + half * 8;
  v8bf lo = *(const v8bf*)(p);
  v8bf hi = *(const v8bf*)(p + 16);
  v16bf r;
#pragma unroll
  for (int i = 0; i < 8; ++i) { r[i] = lo[i]; r[i + 8] = hi[i]; }
  return r;
}

static __device__ __forceinline__ v16bf load_frag_b(const __bf16* __restrict__ base, int ld) {
  const int lane = threadIdx.x & 31;
  const int col  = lane & 15;
  const int half = lane >> 4;
  const __bf16* p = base + (size_t)col * ld + half * 8;
  v8bf lo = *(const v8bf*)(p);
  v8bf hi = *(const v8bf*)(p + 16);
  v16bf r;
#pragma unroll
  for (int i = 0; i < 8; ++i) { r[i] = lo[i]; r[i + 8] = hi[i]; }
  return r;
}

// ---------------------------------------------------------------------------
// Problem constants
// ---------------------------------------------------------------------------
#define Bb   8
#define Ss   2048
#define Dd   1024
#define Mm   (Bb * Ss)   // 16384

// ---------------------------------------------------------------------------
// Convert x (f32) -> bf16, vectorized 4-wide
// ---------------------------------------------------------------------------
__global__ __launch_bounds__(256) void cvt_x_kernel(const float* __restrict__ x,
                                                    __bf16* __restrict__ xb, int n4) {
  for (int i = blockIdx.x * blockDim.x + threadIdx.x; i < n4; i += gridDim.x * blockDim.x) {
    v4f v = *(const v4f*)(x + (size_t)i * 4);
    __bf16* o = xb + (size_t)i * 4;
    o[0] = (__bf16)v[0]; o[1] = (__bf16)v[1]; o[2] = (__bf16)v[2]; o[3] = (__bf16)v[3];
  }
}

// Convert + transpose kernel weights: kT[j][n][k] = kernel[j][k][n]  (bf16)
__global__ __launch_bounds__(256) void cvt_kT_kernel(const float* __restrict__ kern,
                                                     __bf16* __restrict__ kT) {
  const int j = blockIdx.z;
  const size_t base = (size_t)j * Dd * Dd;
  for (int i = blockIdx.x * blockDim.x + threadIdx.x; i < Dd * Dd; i += gridDim.x * blockDim.x) {
    int k = i >> 10;         // row of kernel
    int n = i & 1023;        // col of kernel
    kT[base + (size_t)n * Dd + k] = (__bf16)kern[base + i];
  }
}

// ---------------------------------------------------------------------------
// Projection GEMM: W_j = x @ kernel[j]   (bf16 in, f32 acc, bf16 out)
// Workgroup = 128 threads = 4 waves (2x2), tile 128x128; each wave 64x64
// (4x4 fragments -> each loaded fragment feeds 4 WMMAs).
// j==0 -> wq row-major, j==1 -> wk row-major, j==2 -> wvT transposed [Dd][Mm]
// ---------------------------------------------------------------------------
__global__ __launch_bounds__(128) void proj_kernel(const __bf16* __restrict__ xb,
                                                   const __bf16* __restrict__ kT,
                                                   __bf16* __restrict__ wq,
                                                   __bf16* __restrict__ wk,
                                                   __bf16* __restrict__ wvT) {
  const int j    = blockIdx.z;
  const int wave = threadIdx.x >> 5;
  const int wm   = wave >> 1, wn = wave & 1;
  const int m0   = blockIdx.x * 128 + wm * 64;
  const int n0   = blockIdx.y * 128 + wn * 64;

  const __bf16* A  = xb + (size_t)m0 * Dd;
  const __bf16* Bt = kT + (size_t)j * Dd * Dd + (size_t)n0 * Dd;

  v8f acc[4][4] = {};
  for (int k0 = 0; k0 < Dd; k0 += 32) {
    v16bf af[4], bf[4];
#pragma unroll
    for (int i = 0; i < 4; ++i) af[i] = load_frag_a(A + (size_t)(16 * i) * Dd + k0, Dd);
#pragma unroll
    for (int i = 0; i < 4; ++i) bf[i] = load_frag_b(Bt + (size_t)(16 * i) * Dd + k0, Dd);
#pragma unroll
    for (int mi = 0; mi < 4; ++mi)
#pragma unroll
      for (int ni = 0; ni < 4; ++ni) acc[mi][ni] = WMMA_BF16(af[mi], bf[ni], acc[mi][ni]);
  }

  const int lane = threadIdx.x & 31;
  const int cn   = lane & 15;
  const int half = lane >> 4;
  if (j < 2) {
    __bf16* out = (j == 0) ? wq : wk;
#pragma unroll
    for (int mi = 0; mi < 4; ++mi)
#pragma unroll
      for (int ni = 0; ni < 4; ++ni)
#pragma unroll
        for (int r = 0; r < 8; ++r) {
          int m = m0 + mi * 16 + half * 8 + r;
          int n = n0 + ni * 16 + cn;
          out[(size_t)m * Dd + n] = (__bf16)acc[mi][ni][r];
        }
  } else {
    // wvT[n][m]: per lane, 8 consecutive m -> one packed 16B store
#pragma unroll
    for (int mi = 0; mi < 4; ++mi)
#pragma unroll
      for (int ni = 0; ni < 4; ++ni) {
        int n  = n0 + ni * 16 + cn;
        int mb = m0 + mi * 16 + half * 8;
        v8bf v;
#pragma unroll
        for (int r = 0; r < 8; ++r) v[r] = (__bf16)acc[mi][ni][r];
        *(v8bf*)(wvT + (size_t)n * Mm + mb) = v;
      }
  }
}

// ---------------------------------------------------------------------------
// QK^T + softmax-over-batch, fused.
// Workgroup = 256 threads = 8 waves; wave w computes the full 64x64 score
// tile for batch b=w (16 WMMA fragments, f32 acc). Tiles go to dynamic LDS
// (8*64*64 f32 = 128KB of the WGP's 320KB), softmax runs across b in LDS,
// attn written out as bf16.
// ---------------------------------------------------------------------------
__global__ __launch_bounds__(256) void qk_softmax_kernel(const __bf16* __restrict__ wq,
                                                         const __bf16* __restrict__ wk,
                                                         __bf16* __restrict__ attn) {
  extern __shared__ float smem[];  // [8][64][64]
  const int q0 = blockIdx.x * 64;
  const int k0 = blockIdx.y * 64;
  const int b  = threadIdx.x >> 5;  // wave id == batch

  const __bf16* A  = wq + ((size_t)b * Ss + q0) * Dd;
  const __bf16* Bt = wk + ((size_t)b * Ss + k0) * Dd;  // B col n = wk row (k0+n), K-contig

  v8f acc[4][4] = {};
  for (int kk = 0; kk < Dd; kk += 32) {
    v16bf af[4], bf[4];
#pragma unroll
    for (int i = 0; i < 4; ++i) af[i] = load_frag_a(A + (size_t)(16 * i) * Dd + kk, Dd);
#pragma unroll
    for (int i = 0; i < 4; ++i) bf[i] = load_frag_b(Bt + (size_t)(16 * i) * Dd + kk, Dd);
#pragma unroll
    for (int mi = 0; mi < 4; ++mi)
#pragma unroll
      for (int ni = 0; ni < 4; ++ni) acc[mi][ni] = WMMA_BF16(af[mi], bf[ni], acc[mi][ni]);
  }

  const float scale = 0.03125f;  // 1/sqrt(1024)
  const int lane = threadIdx.x & 31;
  const int cn   = lane & 15;
  const int half = lane >> 4;
  float* sb = smem + b * 4096;
#pragma unroll
  for (int mi = 0; mi < 4; ++mi)
#pragma unroll
    for (int ni = 0; ni < 4; ++ni)
#pragma unroll
      for (int r = 0; r < 8; ++r) {
        int m = mi * 16 + half * 8 + r;
        int n = ni * 16 + cn;
        sb[m * 64 + n] = acc[mi][ni][r] * scale;
      }
  __syncthreads();

  // softmax over the 8 batches for each of the 4096 (q,k) positions
  for (int i = 0; i < 16; ++i) {
    int pos = threadIdx.x + i * 256;
    float mx = -3.0e38f;
#pragma unroll
    for (int bb = 0; bb < 8; ++bb) mx = fmaxf(mx, smem[bb * 4096 + pos]);
    float e[8];
    float sum = 0.f;
#pragma unroll
    for (int bb = 0; bb < 8; ++bb) { e[bb] = __expf(smem[bb * 4096 + pos] - mx); sum += e[bb]; }
    float inv = __frcp_rn(sum);
    int q = pos >> 6, k = pos & 63;
#pragma unroll
    for (int bb = 0; bb < 8; ++bb)
      attn[((size_t)bb * Ss + q0 + q) * Ss + k0 + k] = (__bf16)(e[bb] * inv);
  }
}

// ---------------------------------------------------------------------------
// Output GEMM: out_b = attn_b @ wv_b   (bf16 in, f32 out)
// Workgroup = 128 threads = 4 waves (2x2), tile 128x128 over (q, e); each
// wave 64x64 (4x4 fragments); grid.z = b.
// ---------------------------------------------------------------------------
__global__ __launch_bounds__(128) void out_kernel(const __bf16* __restrict__ attn,
                                                  const __bf16* __restrict__ wvT,
                                                  float* __restrict__ out) {
  const int b    = blockIdx.z;
  const int wave = threadIdx.x >> 5;
  const int wm   = wave >> 1, wn = wave & 1;
  const int q0   = blockIdx.x * 128 + wm * 64;
  const int e0   = blockIdx.y * 128 + wn * 64;

  const __bf16* A  = attn + ((size_t)b * Ss + q0) * Ss;            // ld = Ss
  const __bf16* Bt = wvT + (size_t)e0 * Mm + (size_t)b * Ss;       // col e, K-contig, ld = Mm

  v8f acc[4][4] = {};
  for (int k0 = 0; k0 < Ss; k0 += 32) {
    v16bf af[4], bf[4];
#pragma unroll
    for (int i = 0; i < 4; ++i) af[i] = load_frag_a(A + (size_t)(16 * i) * Ss + k0, Ss);
#pragma unroll
    for (int i = 0; i < 4; ++i) bf[i] = load_frag_b(Bt + (size_t)(16 * i) * Mm + k0, Mm);
#pragma unroll
    for (int mi = 0; mi < 4; ++mi)
#pragma unroll
      for (int ni = 0; ni < 4; ++ni) acc[mi][ni] = WMMA_BF16(af[mi], bf[ni], acc[mi][ni]);
  }

  const int lane = threadIdx.x & 31;
  const int cn   = lane & 15;
  const int half = lane >> 4;
#pragma unroll
  for (int mi = 0; mi < 4; ++mi)
#pragma unroll
    for (int ni = 0; ni < 4; ++ni)
#pragma unroll
      for (int r = 0; r < 8; ++r) {
        int q = q0 + mi * 16 + half * 8 + r;
        int e = e0 + ni * 16 + cn;
        out[((size_t)b * Ss + q) * Dd + e] = acc[mi][ni][r];
      }
}

// ---------------------------------------------------------------------------
// Host launcher
// ---------------------------------------------------------------------------
extern "C" void kernel_launch(void* const* d_in, const int* in_sizes, int n_in,
                              void* d_out, int out_size, void* d_ws, size_t ws_size,
                              hipStream_t stream) {
  const float* x    = (const float*)d_in[0];   // [8,2048,1024]
  const float* kern = (const float*)d_in[1];   // [3,1024,1024]
  float*       out  = (float*)d_out;           // [8,2048,1024]

  char* ws = (char*)d_ws;
  size_t off = 0;
  __bf16* xb   = (__bf16*)(ws + off); off += (size_t)Mm * Dd * 2;        // 32 MB
  __bf16* kT   = (__bf16*)(ws + off); off += (size_t)3 * Dd * Dd * 2;    //  6 MB
  __bf16* wq   = (__bf16*)(ws + off); off += (size_t)Mm * Dd * 2;        // 32 MB
  __bf16* wk   = (__bf16*)(ws + off); off += (size_t)Mm * Dd * 2;        // 32 MB
  __bf16* wvT  = (__bf16*)(ws + off); off += (size_t)Dd * Mm * 2;        // 32 MB
  __bf16* attn = (__bf16*)(ws + off); off += (size_t)Bb * Ss * Ss * 2;   // 64 MB

  // 1) convert inputs to bf16 (weights also transposed for K-contiguous B frags)
  cvt_x_kernel<<<4096, 256, 0, stream>>>(x, xb, (Mm * Dd) / 4);
  cvt_kT_kernel<<<dim3(1024, 1, 3), 256, 0, stream>>>(kern, kT);

  // 2) projections: wq, wk (row-major), wvT (transposed)
  proj_kernel<<<dim3(Mm / 128, Dd / 128, 3), 128, 0, stream>>>(xb, kT, wq, wk, wvT);

  // 3) fused QK^T + softmax over batch axis (128 KB dynamic LDS per WG)
  qk_softmax_kernel<<<dim3(Ss / 64, Ss / 64), 256, 8 * 64 * 64 * (int)sizeof(float), stream>>>(
      wq, wk, attn);

  // 4) out = attn @ wv
  out_kernel<<<dim3(Ss / 128, Dd / 128, Bb), 128, 0, stream>>>(attn, wvT, out);
}
